// SpectralTimeModulatedSFNO_38405597561183
// MI455X (gfx1250) — matrix operand order repaired and checked by reading, not compiled, not run on hardware
//
#include <hip/hip_runtime.h>
#include <math.h>

#define DPI 3.14159265358979323846

typedef __bf16 bf16_t;
typedef __attribute__((ext_vector_type(8)))  __bf16 v8bf;
typedef __attribute__((ext_vector_type(16))) __bf16 v16bf;
typedef __attribute__((ext_vector_type(8)))  float  v8f;

// D = A x B + C, A 16x32 bf16, B 32x16 bf16, C/D 16x16 f32
#define WMMA_BF16(a, b, c) \
  __builtin_amdgcn_wmma_f32_16x16x32_bf16(false, (a), false, (b), (short)0, (c), false, false)

// ---------------------------------------------------------------------------
// Fragment loaders (wave32 layouts per CDNA5 ISA 7.12.2)
// A (16x32, M x K), row-major [rows][ld]:
//   lane 0-15 : row = lane, K = {k..k+7, k+16..k+23}
//   lane 16-31: row = lane-15? no -> row = lane&15, K = {k+8..k+15, k+24..k+31}
__device__ __forceinline__ v16bf frag_a(const bf16_t* __restrict__ A, int row,
                                        int ld, int k, int lane) {
  const bf16_t* p = A + (size_t)row * ld + k + ((lane >> 4) << 3);
  v8bf c0 = *(const v8bf*)(p);
  v8bf c1 = *(const v8bf*)(p + 16);
  return __builtin_shufflevector(c0, c1, 0, 1, 2, 3, 4, 5, 6, 7,
                                 8, 9, 10, 11, 12, 13, 14, 15);
}
// B (32x16, K x N) fetched from Bt = B^T row-major [N][ld]:
//   lanes 0-15 hold K = {k..k+7, k+16..k+23}, lanes 16-31 hold K = {k+8..k+15, k+24..k+31}
__device__ __forceinline__ v16bf frag_b(const bf16_t* __restrict__ Bt, int col,
                                        int ld, int k, int lane) {
  const bf16_t* p = Bt + (size_t)col * ld + k + ((lane >> 4) << 3);
  v8bf c0 = *(const v8bf*)(p);
  v8bf c1 = *(const v8bf*)(p + 16);
  return __builtin_shufflevector(c0, c1, 0, 1, 2, 3, 4, 5, 6, 7,
                                 8, 9, 10, 11, 12, 13, 14, 15);
}

// Wave computes a 16(M) x 64(N) tile of D = A(MxK) * B(KxN)
__device__ __forceinline__ void wave_gemm(const bf16_t* __restrict__ A, int lda,
                                          int arow,
                                          const bf16_t* __restrict__ Bt, int ldb,
                                          int K, int lane, v8f acc[4]) {
  for (int k = 0; k < K; k += 32) {
    v16bf a = frag_a(A, arow, lda, k, lane);
#pragma unroll
    for (int j = 0; j < 4; ++j) {
      v16bf b = frag_b(Bt, (lane & 15) + j * 16, ldb, k, lane);
      acc[j] = WMMA_BF16(a, b, acc[j]);
    }
  }
}

// ---------------------------------------------------------------------------
// Setup kernels
// ---------------------------------------------------------------------------

__global__ void k_zero_bf(bf16_t* p, size_t n) {
  size_t i = (size_t)blockIdx.x * blockDim.x + threadIdx.x;
  size_t st = (size_t)gridDim.x * blockDim.x;
  for (; i < n; i += st) p[i] = (bf16_t)0.0f;
}

// Gauss-Legendre nodes (descending) + weights for n=256, Newton in f64
__global__ void k_gl(double* xg, double* wg) {
  const int n = 256;
  int i = threadIdx.x;
  double x = cos(DPI * (i + 0.75) / (n + 0.5));
  for (int it = 0; it < 10; ++it) {
    double p0 = 1.0, p1 = x;
    for (int k = 2; k <= n; ++k) {
      double pk = ((2.0 * k - 1.0) * x * p1 - (k - 1.0) * p0) / k;
      p0 = p1; p1 = pk;
    }
    double dp = n * (x * p1 - p0) / (x * x - 1.0);
    x -= p1 / dp;
  }
  double p0 = 1.0, p1 = x;
  for (int k = 2; k <= n; ++k) {
    double pk = ((2.0 * k - 1.0) * x * p1 - (k - 1.0) * p0) / k;
    p0 = p1; p1 = pk;
  }
  double dp = n * (x * p1 - p0) / (x * x - 1.0);
  xg[i] = x;
  wg[i] = 2.0 / ((1.0 - x * x) * dp * dp);
}

// Normalized associated Legendre tables, l,m < 128, t < 256.
// PW[m][l][t] = P_l^m(x_t) * w_t  (forward, A row-major over t)
// Ptl[m][t][l] = P_l^m(x_t)        (inverse, A row-major over l)
__global__ void k_leg(const double* __restrict__ xg, const double* __restrict__ wg,
                      bf16_t* __restrict__ PW, bf16_t* __restrict__ Ptl) {
  int t = threadIdx.x;  // 256 threads
  double x = xg[t], w = wg[t];
  double sx = sqrt(fmax(0.0, 1.0 - x * x));
  double pmm = 0.28209479177387814;  // 1/sqrt(4*pi)
  for (int m = 0; m < 128; ++m) {
    if (m > 0) pmm *= -sqrt((2.0 * m + 1.0) / (2.0 * m)) * sx;
    double plm2 = pmm;
    PW [((size_t)m * 128 + m) * 256 + t] = (bf16_t)(float)(pmm * w);
    Ptl[((size_t)m * 256 + t) * 128 + m] = (bf16_t)(float)pmm;
    if (m + 1 < 128) {
      double plm1 = sqrt(2.0 * m + 3.0) * x * pmm;
      PW [((size_t)m * 128 + (m + 1)) * 256 + t] = (bf16_t)(float)(plm1 * w);
      Ptl[((size_t)m * 256 + t) * 128 + (m + 1)] = (bf16_t)(float)plm1;
      for (int l = m + 2; l < 128; ++l) {
        double a = sqrt((4.0 * (double)l * l - 1.0) /
                        ((double)l * l - (double)m * m));
        double b = sqrt((((l - 1.0) * (l - 1.0)) - (double)m * m) /
                        (4.0 * (l - 1.0) * (l - 1.0) - 1.0));
        double p = a * (x * plm1 - b * plm2);
        PW [((size_t)m * 128 + l) * 256 + t] = (bf16_t)(float)(p * w);
        Ptl[((size_t)m * 256 + t) * 128 + l] = (bf16_t)(float)p;
        plm2 = plm1; plm1 = p;
      }
    }
  }
}

// trigF: [512 rows][512 cols] bf16. rows 0..254: cos(2*pi*p*m'/511)*scale,
// rows 255..509: -sin(...)*scale (m' = r-127 / r-382), pads zero. cols = p.
// trigI: [512 p][512 k] bf16. k 0..254: cos, k 255..509: -sin (no scale).
__global__ void k_trig(bf16_t* __restrict__ trigF, bf16_t* __restrict__ trigI) {
  int r = blockIdx.x;  // 0..511
  const double sc = 2.0 * DPI / 511.0;
  for (int c = threadIdx.x; c < 512; c += blockDim.x) {
    float vF = 0.0f;
    if (r < 510 && c < 511) {
      int mp = (r < 255) ? (r - 127) : (r - 255 - 127);
      double ang = 2.0 * DPI * (double)c * (double)mp / 511.0;
      vF = (float)(((r < 255) ? cos(ang) : -sin(ang)) * sc);
    }
    trigF[(size_t)r * 512 + c] = (bf16_t)vF;
    float vI = 0.0f;
    if (r < 511 && c < 510) {
      int mp = (c < 255) ? (c - 127) : (c - 255 - 127);
      double ang = 2.0 * DPI * (double)r * (double)mp / 511.0;
      vI = (float)((c < 255) ? cos(ang) : -sin(ang));
    }
    trigI[(size_t)r * 512 + c] = (bf16_t)vI;
  }
}

// x (t,p,c) f32  ->  xT (t,c,512p) bf16 (p=511 padded zero)
__global__ void k_convx(const float* __restrict__ x, bf16_t* __restrict__ xT) {
  size_t idx = (size_t)blockIdx.x * blockDim.x + threadIdx.x;
  if (idx >= (size_t)256 * 64 * 512) return;
  int p = idx & 511;
  int c = (idx >> 9) & 63;
  int t = (int)(idx >> 15);
  float v = (p < 511) ? x[((size_t)t * 511 + p) * 64 + c] : 0.0f;
  xT[idx] = (bf16_t)v;
}

__global__ void k_convR(const float* __restrict__ Rre, const float* __restrict__ Rim,
                        bf16_t* __restrict__ Rr, bf16_t* __restrict__ Ri,
                        bf16_t* __restrict__ Rin) {
  size_t idx = (size_t)blockIdx.x * blockDim.x + threadIdx.x;
  if (idx >= (size_t)128 * 64 * 64) return;
  Rr[idx] = (bf16_t)Rre[idx];
  Ri[idx] = (bf16_t)Rim[idx];
  Rin[idx] = (bf16_t)(-Rim[idx]);
}

__global__ void k_phi(const float* __restrict__ Are, const float* __restrict__ Aim,
                      const float* __restrict__ te, float* __restrict__ pr,
                      float* __restrict__ pi) {
  int l = threadIdx.x;  // 128
  if (l >= 128) return;
  float sr = 0.f, si = 0.f;
  for (int c = 0; c < 256; ++c) {
    float t = te[c];
    sr += Are[l * 256 + c] * t;
    si += Aim[l * 256 + c] * t;
  }
  pr[l] = sr;
  pi[l] = si;
}

// ---------------------------------------------------------------------------
// Stage A: F[t] (510m x 64c) = trigF (510x512) * xT[t]^T ; rows<255 -> Fr, else Fi
// Fr/Fi layout: [m][c][t]
// ---------------------------------------------------------------------------
__global__ __launch_bounds__(256) void k_stageA(const bf16_t* __restrict__ trigF,
                                                const bf16_t* __restrict__ xT,
                                                bf16_t* __restrict__ Fr,
                                                bf16_t* __restrict__ Fi) {
  int lane = threadIdx.x & 31, wave = threadIdx.x >> 5;
  int t = blockIdx.y;
  int mbase = blockIdx.x * 128 + wave * 16;
  const bf16_t* Bt = xT + (size_t)t * 64 * 512;
  v8f acc[4] = {};
  wave_gemm(trigF, 512, mbase + (lane & 15), Bt, 512, 512, lane, acc);
  int lh = lane >> 4, n0 = lane & 15;
#pragma unroll
  for (int j = 0; j < 4; ++j)
#pragma unroll
    for (int v = 0; v < 8; ++v) {
      int row = mbase + v + lh * 8;
      int n = n0 + j * 16;
      if (row < 255)
        Fr[((size_t)row * 64 + n) * 256 + t] = (bf16_t)acc[j][v];
      else if (row < 510)
        Fi[((size_t)(row - 255) * 64 + n) * 256 + t] = (bf16_t)acc[j][v];
    }
}

// ---------------------------------------------------------------------------
// Stage B: G[:,mi,:] (128l x 64c) = sgn * PW[|m'|] (128x256t) * F[:,mi,:]
// G layout: [l][256 m][c]  (row m=255 is pad, masked downstream)
// ---------------------------------------------------------------------------
__global__ __launch_bounds__(256) void k_stageB(const bf16_t* __restrict__ PW,
                                                const bf16_t* __restrict__ Fr,
                                                const bf16_t* __restrict__ Fi,
                                                bf16_t* __restrict__ Gr,
                                                bf16_t* __restrict__ Gi) {
  int lane = threadIdx.x & 31, wave = threadIdx.x >> 5;
  int mi = blockIdx.x;  // 0..254
  int ri = blockIdx.y;  // 0 = real, 1 = imag
  int mp = mi - 127, am = (mp < 0) ? -mp : mp;
  float sgn = (mp < 0 && (am & 1)) ? -1.0f : 1.0f;
  const bf16_t* A = PW + (size_t)am * 128 * 256;
  const bf16_t* Bt = (ri ? Fi : Fr) + (size_t)mi * 64 * 256;
  bf16_t* G = ri ? Gi : Gr;
  int lbase = wave * 16;
  v8f acc[4] = {};
  wave_gemm(A, 256, lbase + (lane & 15), Bt, 256, 256, lane, acc);
  int lh = lane >> 4, n0 = lane & 15;
#pragma unroll
  for (int j = 0; j < 4; ++j)
#pragma unroll
    for (int v = 0; v < 8; ++v) {
      int l = lbase + v + lh * 8;
      int n = n0 + j * 16;
      G[((size_t)l * 256 + mi) * 64 + n] = (bf16_t)(sgn * acc[j][v]);
    }
}

// ---------------------------------------------------------------------------
// Stage C: per l: O = (Gr + iGi)(Rr + iRi)^T * phi_c[l]
//   Or = Gr*Rr^T + Gi*(-Ri)^T ; Oi = Gr*Ri^T + Gi*Rr^T (bf16 NEG not avail -> Rin)
// Output layout: [m][o][l]
// ---------------------------------------------------------------------------
__global__ __launch_bounds__(256) void k_stageC(const bf16_t* __restrict__ Gr,
                                                const bf16_t* __restrict__ Gi,
                                                const bf16_t* __restrict__ Rr,
                                                const bf16_t* __restrict__ Ri,
                                                const bf16_t* __restrict__ Rin,
                                                const float* __restrict__ pr,
                                                const float* __restrict__ pi,
                                                bf16_t* __restrict__ OrS,
                                                bf16_t* __restrict__ OiS) {
  int lane = threadIdx.x & 31, wave = threadIdx.x >> 5;
  int l = blockIdx.y;
  int mbase = blockIdx.x * 128 + wave * 16;
  const bf16_t* Ar = Gr + (size_t)l * 256 * 64;
  const bf16_t* Ai = Gi + (size_t)l * 256 * 64;
  const bf16_t* Br = Rr + (size_t)l * 64 * 64;
  const bf16_t* Bi = Ri + (size_t)l * 64 * 64;
  const bf16_t* Bn = Rin + (size_t)l * 64 * 64;
  v8f aR[4] = {}, aI[4] = {};
  int arow = mbase + (lane & 15);
  for (int k = 0; k < 64; k += 32) {
    v16bf fr = frag_a(Ar, arow, 64, k, lane);
    v16bf fi = frag_a(Ai, arow, 64, k, lane);
#pragma unroll
    for (int j = 0; j < 4; ++j) {
      int n = (lane & 15) + j * 16;
      v16bf br = frag_b(Br, n, 64, k, lane);
      v16bf bi = frag_b(Bi, n, 64, k, lane);
      v16bf bn = frag_b(Bn, n, 64, k, lane);
      aR[j] = WMMA_BF16(fr, br, aR[j]);
      aR[j] = WMMA_BF16(fi, bn, aR[j]);
      aI[j] = WMMA_BF16(fr, bi, aI[j]);
      aI[j] = WMMA_BF16(fi, br, aI[j]);
    }
  }
  float p_r = pr[l], p_i = pi[l];
  int lh = lane >> 4, n0 = lane & 15;
#pragma unroll
  for (int j = 0; j < 4; ++j)
#pragma unroll
    for (int v = 0; v < 8; ++v) {
      int m = mbase + v + lh * 8;
      if (m >= 255) continue;
      int n = n0 + j * 16;
      float orv = aR[j][v], oiv = aI[j][v];
      OrS[((size_t)m * 64 + n) * 128 + l] = (bf16_t)(p_r * orv - p_i * oiv);
      OiS[((size_t)m * 64 + n) * 128 + l] = (bf16_t)(p_r * oiv + p_i * orv);
    }
}

// ---------------------------------------------------------------------------
// Stage D: H[:,mi,:] (256t x 64c) = sgn * Ptl[|m'|] (256x128l) * O[:,mi,:]
// Hst layout: [t][c][512], real at col mi, imag at col 255+mi, cols 510/511 = 0
// ---------------------------------------------------------------------------
__global__ __launch_bounds__(256) void k_stageD(const bf16_t* __restrict__ Ptl,
                                                const bf16_t* __restrict__ OrS,
                                                const bf16_t* __restrict__ OiS,
                                                bf16_t* __restrict__ Hst) {
  int lane = threadIdx.x & 31, wave = threadIdx.x >> 5;
  int mi = blockIdx.x;  // 0..254
  int tt = blockIdx.y;  // 0..1 (t tiles of 128)
  int ri = blockIdx.z;  // 0 real, 1 imag
  int mp = mi - 127, am = (mp < 0) ? -mp : mp;
  float sgn = (mp < 0 && (am & 1)) ? -1.0f : 1.0f;
  const bf16_t* A = Ptl + (size_t)am * 256 * 128;
  const bf16_t* Bt = (ri ? OiS : OrS) + (size_t)mi * 64 * 128;
  int tbase = tt * 128 + wave * 16;
  v8f acc[4] = {};
  wave_gemm(A, 128, tbase + (lane & 15), Bt, 128, 128, lane, acc);
  int lh = lane >> 4, n0 = lane & 15;
  int col = mi + ri * 255;
#pragma unroll
  for (int j = 0; j < 4; ++j)
#pragma unroll
    for (int v = 0; v < 8; ++v) {
      int t = tbase + v + lh * 8;
      int n = n0 + j * 16;
      Hst[((size_t)t * 64 + n) * 512 + col] = (bf16_t)(sgn * acc[j][v]);
    }
}

// ---------------------------------------------------------------------------
// Stage E: y[t] (511p x 64c) = trigI (512x512) * Hst[t]^T  -> f32 output
// ---------------------------------------------------------------------------
__global__ __launch_bounds__(256) void k_stageE(const bf16_t* __restrict__ trigI,
                                                const bf16_t* __restrict__ Hst,
                                                float* __restrict__ out) {
  int lane = threadIdx.x & 31, wave = threadIdx.x >> 5;
  int t = blockIdx.y;
  int pbase = blockIdx.x * 128 + wave * 16;
  const bf16_t* Bt = Hst + (size_t)t * 64 * 512;
  v8f acc[4] = {};
  wave_gemm(trigI, 512, pbase + (lane & 15), Bt, 512, 512, lane, acc);
  int lh = lane >> 4, n0 = lane & 15;
#pragma unroll
  for (int j = 0; j < 4; ++j)
#pragma unroll
    for (int v = 0; v < 8; ++v) {
      int p = pbase + v + lh * 8;
      if (p >= 511) continue;
      int n = n0 + j * 16;
      out[((size_t)t * 511 + p) * 64 + n] = acc[j][v];
    }
}

// ---------------------------------------------------------------------------
extern "C" void kernel_launch(void* const* d_in, const int* in_sizes, int n_in,
                              void* d_out, int out_size, void* d_ws, size_t ws_size,
                              hipStream_t stream) {
  const float* x      = (const float*)d_in[0];  // (256, 511, 64)
  const float* t_emb  = (const float*)d_in[1];  // (256,)
  const float* A_real = (const float*)d_in[2];  // (128, 256)
  const float* A_imag = (const float*)d_in[3];  // (128, 256)
  const float* R_real = (const float*)d_in[4];  // (128, 64, 64)
  const float* R_imag = (const float*)d_in[5];  // (128, 64, 64)
  float* out = (float*)d_out;                   // (256, 511, 64)

  char* base = (char*)d_ws;
  size_t off = 0;
  auto carve = [&](size_t bytes) -> void* {
    off = (off + 255) & ~(size_t)255;
    void* p = base + off;
    off += bytes;
    return p;
  };
  double* xg    = (double*)carve(256 * 8);
  double* wg    = (double*)carve(256 * 8);
  const size_t nPW = (size_t)128 * 128 * 256;
  bf16_t* PW    = (bf16_t*)carve(nPW * 2);
  bf16_t* Ptl   = (bf16_t*)carve(nPW * 2);
  bf16_t* trigF = (bf16_t*)carve((size_t)512 * 512 * 2);
  bf16_t* trigI = (bf16_t*)carve((size_t)512 * 512 * 2);
  bf16_t* xT    = (bf16_t*)carve((size_t)256 * 64 * 512 * 2);
  bf16_t* Fr    = (bf16_t*)carve((size_t)255 * 64 * 256 * 2);
  bf16_t* Fi    = (bf16_t*)carve((size_t)255 * 64 * 256 * 2);
  bf16_t* Gr    = (bf16_t*)carve((size_t)128 * 256 * 64 * 2);
  bf16_t* Gi    = (bf16_t*)carve((size_t)128 * 256 * 64 * 2);
  bf16_t* Rr    = (bf16_t*)carve((size_t)128 * 64 * 64 * 2);
  bf16_t* Ri    = (bf16_t*)carve((size_t)128 * 64 * 64 * 2);
  bf16_t* Rin   = (bf16_t*)carve((size_t)128 * 64 * 64 * 2);
  float*  pr    = (float*)carve(128 * 4);
  float*  pi    = (float*)carve(128 * 4);
  bf16_t* OrS   = (bf16_t*)carve((size_t)255 * 64 * 128 * 2);
  bf16_t* OiS   = (bf16_t*)carve((size_t)255 * 64 * 128 * 2);
  const size_t nH = (size_t)256 * 64 * 512;
  bf16_t* Hst   = (bf16_t*)carve(nH * 2);
  (void)ws_size; (void)in_sizes; (void)n_in; (void)out_size;

  // tables that are only partially written must be exact zero elsewhere
  k_zero_bf<<<1024, 256, 0, stream>>>(PW, nPW);
  k_zero_bf<<<1024, 256, 0, stream>>>(Ptl, nPW);
  k_zero_bf<<<1024, 256, 0, stream>>>(Gr, (size_t)128 * 256 * 64);
  k_zero_bf<<<1024, 256, 0, stream>>>(Gi, (size_t)128 * 256 * 64);
  k_zero_bf<<<1024, 256, 0, stream>>>(Hst, nH);

  k_gl<<<1, 256, 0, stream>>>(xg, wg);
  k_leg<<<1, 256, 0, stream>>>(xg, wg, PW, Ptl);
  k_trig<<<512, 256, 0, stream>>>(trigF, trigI);
  k_convx<<<(unsigned)(((size_t)256 * 64 * 512 + 255) / 256), 256, 0, stream>>>(x, xT);
  k_convR<<<(unsigned)(((size_t)128 * 64 * 64 + 255) / 256), 256, 0, stream>>>(
      R_real, R_imag, Rr, Ri, Rin);
  k_phi<<<1, 128, 0, stream>>>(A_real, A_imag, t_emb, pr, pi);

  k_stageA<<<dim3(4, 256), 256, 0, stream>>>(trigF, xT, Fr, Fi);
  k_stageB<<<dim3(255, 2), 256, 0, stream>>>(PW, Fr, Fi, Gr, Gi);
  k_stageC<<<dim3(2, 128), 256, 0, stream>>>(Gr, Gi, Rr, Ri, Rin, pr, pi, OrS, OiS);
  k_stageD<<<dim3(255, 2, 2), 256, 0, stream>>>(Ptl, OrS, OiS, Hst);
  k_stageE<<<dim3(4, 256), 256, 0, stream>>>(trigI, Hst, out);
}
